// BahdanauAttentionCell_80470507258063
// MI455X (gfx1250) — matrix operand order repaired and checked by reading, not compiled, not run on hardware
//
#include <hip/hip_runtime.h>
#include <hip/hip_bf16.h>
#include <cstddef>

typedef __attribute__((ext_vector_type(16))) _Float16 v16h;
typedef __attribute__((ext_vector_type(8)))  float    v8f;

#define BATCH 64
#define TLEN  2000
#define ADIM  256
#define FDIM  32
#define KW    31
#define MAXS  2000
#define MASK_VALF (-4294967295.0f)   // -2^32 + 1, rounds to -2^32 in f32 (same as reference)

// Hardware tanh (CDNA5 V_TANH_F32 TRANS op). Fallback chain keeps it compiling
// on any toolchain; the asm path adds v_nop for the TRANS result-use hazard.
static __device__ inline float hw_tanh(float x) {
#if __has_builtin(__builtin_amdgcn_tanhf)
    return __builtin_amdgcn_tanhf(x);
#elif __has_builtin(__builtin_amdgcn_tanh_f32)
    return __builtin_amdgcn_tanh_f32(x);
#else
    float y;
    asm("v_tanh_f32 %0, %1\n\tv_nop" : "=v"(y) : "v"(x));
    return y;
#endif
}

// Hardware exp via v_exp_f32 (2^x): exp(x) = 2^(x*log2(e)).
static __device__ inline float hw_exp(float x) {
#if __has_builtin(__builtin_amdgcn_exp2f)
    return __builtin_amdgcn_exp2f(x * 1.4426950408889634f);
#else
    return expf(x);
#endif
}

// ---------------------------------------------------------------------------
// Kernel 1: fused conv -> WMMA location GEMM -> tanh energy
// grid = (BATCH, 16), block = 256 threads (8 waves). Wave w handles rows
// t0 = blockIdx.y*128 + w*16 .. t0+15 of batch b = blockIdx.x.
// Writes raw (unmasked) energy into energy[b*MAXS + t] (= d_out first half).
// ---------------------------------------------------------------------------
__global__ __launch_bounds__(256)
void bahdanau_energy_kernel(const float* __restrict__ query,     // [B, A]
                            const float* __restrict__ prev_w,    // [B, MAXS]
                            const float* __restrict__ w_memory,  // [B, T, A]
                            const float* __restrict__ conv_w,    // [KW, 1, F]
                            const float* __restrict__ conv_b,    // [F]
                            const float* __restrict__ loc_w,     // [F, A]
                            const float* __restrict__ score_v,   // [A]
                            const float* __restrict__ score_b,   // [A]
                            float* __restrict__ energy)          // [B, MAXS]
{
    __shared__ float    s_cw[KW * FDIM];     // conv weights, f32 (3968 B)
    __shared__ float    s_cb[FDIM];          // conv bias
    __shared__ _Float16 s_lw[FDIM * ADIM];   // loc_w as f16   (16 KB)
    __shared__ float    s_base[ADIM];        // query[b,:] + score_b
    __shared__ float    s_sv[ADIM];          // score_v
    __shared__ float    s_prev[128 + KW + 1];// prev_weights halo window (zero-padded)

    const int tid = threadIdx.x;
    const int b   = blockIdx.x;
    const int c0  = blockIdx.y * 128;        // first t of this block's chunk

    // ---- Stage all small operands into LDS once per block ----
    for (int i = tid; i < KW * FDIM; i += 256) s_cw[i] = conv_w[i];
    for (int i = tid; i < FDIM * ADIM; i += 256) s_lw[i] = (_Float16)loc_w[i];
    if (tid < FDIM) s_cb[tid] = conv_b[tid];
    if (tid < ADIM) {
        s_base[tid] = query[b * ADIM + tid] + score_b[tid];
        s_sv[tid]   = score_v[tid];
    }
    if (tid < 128 + KW + 1) {                // window [c0-15, c0+144]
        const int g = c0 - (KW / 2) + tid;
        s_prev[tid] = (g >= 0 && g < TLEN) ? prev_w[b * MAXS + g] : 0.0f;
    }
    __syncthreads();

    const int wave = tid >> 5;
    const int lane = tid & 31;
    const int t0   = c0 + wave * 16;
    if (t0 >= TLEN) return;                  // uniform per wave; after the barrier

    const int hi = (lane >= 16) ? 1 : 0;     // upper half-wave
    const int r  = lane & 15;                // A-fragment row M (and B-fragment col N)

    // ---- Build A fragment: conv location features, ISA 16-bit A 16x32 layout
    //      lanes 0-15 : element h -> K = (h<8 ? h : h+8)
    //      lanes 16-31: element h -> K = 8 + (h<8 ? h : h+8)
    const int kbase = hi ? 8 : 0;
    const int pbase = wave * 16 + r;         // local index of t-15 in s_prev
    float acc[16];
#pragma unroll
    for (int h = 0; h < 16; ++h) {
        const int Kf = kbase + ((h < 8) ? h : h + 8);
        acc[h] = s_cb[Kf];
    }
#pragma unroll 1
    for (int k = 0; k < KW; ++k) {
        const float p = s_prev[pbase + k];   // halo pre-zeroed, no branch
#pragma unroll
        for (int h = 0; h < 16; ++h) {
            const int Kf = kbase + ((h < 8) ? h : h + 8);
            acc[h] = fmaf(p, s_cw[k * FDIM + Kf], acc[h]);
        }
    }
    v16h afrag;
#pragma unroll
    for (int h = 0; h < 16; ++h) afrag[h] = (_Float16)acc[h];

    // ---- Per-row partial energies (C layout: VGPR v, lane -> M = v + 8*hi)
    float part[8];
#pragma unroll
    for (int v = 0; v < 8; ++v) part[v] = 0.0f;

    const int kbB = hi ? 16 : 0;             // (unused) B uses same K order as A fragment
    // Row base pointer for this lane's w_memory elements (column r of each tile).
    const float* wrow = w_memory + ((size_t)(b * TLEN + t0 + hi * 8)) * ADIM + r;

#pragma unroll 2
    for (int j = 0; j < ADIM / 16; ++j) {    // 16 column tiles over A = 256
        const int a = j * 16 + r;

        // B fragment from LDS (element h -> K = kbase + (h<8 ? h : h+8), N = r)
        v16h bfrag;
#pragma unroll
        for (int h = 0; h < 16; ++h) bfrag[h] = s_lw[(kbase + ((h < 8) ? h : h + 8)) * ADIM + a];

        v8f c = {};
        c = __builtin_amdgcn_wmma_f32_16x16x32_f16(
                /*neg_a=*/false, afrag, /*neg_b=*/false, bfrag,
                /*c_mod=*/(short)0, c, /*reuse_a=*/false, /*reuse_b=*/false);

        // Issue all 8 w_memory loads for this tile before consuming any.
        float wm[8];
#pragma unroll
        for (int v = 0; v < 8; ++v)
            wm[v] = wrow[(size_t)v * ADIM + j * 16];

        const float base = s_base[a];
        const float sv   = s_sv[a];
#pragma unroll
        for (int v = 0; v < 8; ++v) {
            const float x = wm[v] + base + c[v];
            part[v] = fmaf(hw_tanh(x), sv, part[v]);   // TEMPERATURE == 1.0
        }
    }

    // Reduce the N-dimension across the 16 lanes of each half-wave.
#pragma unroll
    for (int v = 0; v < 8; ++v) {
        float s = part[v];
        s += __shfl_xor(s, 1, 32);
        s += __shfl_xor(s, 2, 32);
        s += __shfl_xor(s, 4, 32);
        s += __shfl_xor(s, 8, 32);
        part[v] = s;
    }
    if ((lane & 15) == 0) {                  // lane 0 -> rows t0..t0+7, lane 16 -> t0+8..t0+15
        const int tb = t0 + (hi ? 8 : 0);
#pragma unroll
        for (int v = 0; v < 8; ++v) energy[b * MAXS + tb + v] = part[v];
    }
}

// ---------------------------------------------------------------------------
// Kernel 2: masked softmax over T per batch + cumulative new_weights.
// Energy was staged in out[0 .. B*MAXS); each thread reads its own elements
// into registers before overwriting them, so in-place is race-free.
// out layout: [0, B*MAXS) = alignments, [B*MAXS, 2*B*MAXS) = new_weights.
// ---------------------------------------------------------------------------
__global__ __launch_bounds__(256)
void bahdanau_softmax_kernel(const float* __restrict__ prev_w,   // [B, MAXS]
                             const int*   __restrict__ mem_len,  // [B]
                             float* __restrict__ out)
{
    const int b   = blockIdx.x;
    const int tid = threadIdx.x;
    const int len = mem_len[b];

    __shared__ float red[8];

    float ev[8];
    float lmax = -3.4e38f;
#pragma unroll
    for (int i = 0; i < 8; ++i) {
        const int t = tid + i * 256;
        float e = MASK_VALF;
        if (t < TLEN) {
            e = out[b * MAXS + t];
            if (t >= len) e = MASK_VALF;
        }
        ev[i] = e;
        lmax = fmaxf(lmax, e);
    }
    // block max
    for (int off = 16; off > 0; off >>= 1) lmax = fmaxf(lmax, __shfl_xor(lmax, off, 32));
    if ((tid & 31) == 0) red[tid >> 5] = lmax;
    __syncthreads();
    float bmax = red[0];
#pragma unroll
    for (int w = 1; w < 8; ++w) bmax = fmaxf(bmax, red[w]);
    __syncthreads();

    // exp + block sum
    float lsum = 0.0f;
#pragma unroll
    for (int i = 0; i < 8; ++i) {
        const int t = tid + i * 256;
        if (t < TLEN) {
            ev[i] = hw_exp(ev[i] - bmax);
            lsum += ev[i];
        }
    }
    for (int off = 16; off > 0; off >>= 1) lsum += __shfl_xor(lsum, off, 32);
    if ((tid & 31) == 0) red[tid >> 5] = lsum;
    __syncthreads();
    float bsum = 0.0f;
#pragma unroll
    for (int w = 0; w < 8; ++w) bsum += red[w];
    const float inv = 1.0f / bsum;

#pragma unroll
    for (int i = 0; i < 8; ++i) {
        const int t = tid + i * 256;
        if (t < MAXS) {
            const float al = (t < TLEN) ? ev[i] * inv : 0.0f;   // (pad region empty: T==MAXS)
            out[b * MAXS + t] = al;                              // output
            out[BATCH * MAXS + b * MAXS + t] = al + prev_w[b * MAXS + t];  // new_weights
        }
    }
}

// ---------------------------------------------------------------------------
extern "C" void kernel_launch(void* const* d_in, const int* in_sizes, int n_in,
                              void* d_out, int out_size, void* d_ws, size_t ws_size,
                              hipStream_t stream)
{
    (void)in_sizes; (void)n_in; (void)out_size; (void)d_ws; (void)ws_size;

    const float* query    = (const float*)d_in[0];
    const float* prev_w   = (const float*)d_in[1];
    const float* w_memory = (const float*)d_in[2];
    const int*   mem_len  = (const int*)  d_in[3];
    const float* conv_w   = (const float*)d_in[4];
    const float* conv_b   = (const float*)d_in[5];
    const float* loc_w    = (const float*)d_in[6];
    const float* score_v  = (const float*)d_in[7];
    const float* score_b  = (const float*)d_in[8];
    float* out = (float*)d_out;

    // Kernel 1: 64 batches x 16 chunks of 128 rows; 8 waves/block, 16 rows/wave.
    dim3 g1(BATCH, (TLEN + 127) / 128);
    bahdanau_energy_kernel<<<g1, 256, 0, stream>>>(
        query, prev_w, w_memory, conv_w, conv_b, loc_w, score_v, score_b, out);

    // Kernel 2: one block per batch row.
    bahdanau_softmax_kernel<<<BATCH, 256, 0, stream>>>(prev_w, mem_len, out);
}
